// RiemannFMEdgeSelfUpdate_73710228734535
// MI455X (gfx1250) — matrix operand order repaired and checked by reading, not compiled, not run on hardware
//
#include <hip/hip_runtime.h>
#include <hip/hip_bf16.h>

typedef __attribute__((ext_vector_type(16))) _Float16 v16h;
typedef __attribute__((ext_vector_type(8)))  float    v8f;
typedef __attribute__((ext_vector_type(4)))  unsigned int v4u;
typedef __attribute__((ext_vector_type(8)))  int      v8i;
typedef __attribute__((ext_vector_type(4)))  int      v4i;

#define NN 256
#define DD 128
#define BB 2

// ---- WMMA fragment loaders (contiguous-K 16-bit operand order, wave32) -------------------
// A tile: 16(M) x 32(K). lane: m = lane&15, h = lane>>4.
// dword v of the 16-half fragment holds K = 2*v + 16*h, +1.
static __device__ __forceinline__ v16h lda(const _Float16* p, int ld) {
  const int lane = threadIdx.x & 31;
  const int m = lane & 15, hh = lane >> 4;
  v16h a;
#pragma unroll
  for (int v = 0; v < 8; ++v) {
    const int k0 = 2 * v + 16 * hh;
    a[2 * v]     = p[m * ld + k0];
    a[2 * v + 1] = p[m * ld + k0 + 1];
  }
  return a;
}

// B tile: 32(K) x 16(N) sourced from row-major [N][K] buffer (B^T in memory):
// B[k][n] = p[n*ld + k]. lanes 0-15: K=0..15, lanes 16-31: K=16..31.
static __device__ __forceinline__ v16h ldbt(const _Float16* p, int ld) {
  const int lane = threadIdx.x & 31;
  const int n = lane & 15, hh = lane >> 4;
  v16h b;
#pragma unroll
  for (int v = 0; v < 8; ++v) {
    const int k0 = 2 * v + 16 * hh;
    b[2 * v]     = p[n * ld + k0];
    b[2 * v + 1] = p[n * ld + k0 + 1];
  }
  return b;
}

// B tile: 32(K) x 16(N) sourced from row-major [K][N] buffer: B[k][n] = p[k*ld+n].
static __device__ __forceinline__ v16h ldb(const _Float16* p, int ld) {
  const int lane = threadIdx.x & 31;
  const int n = lane & 15, hh = lane >> 4;
  v16h b;
#pragma unroll
  for (int v = 0; v < 8; ++v) {
    const int k0 = 2 * v + 16 * hh;
    b[2 * v]     = p[k0 * ld + n];
    b[2 * v + 1] = p[(k0 + 1) * ld + n];
  }
  return b;
}

#define WMMA_F16(c, a, b) \
  (c) = __builtin_amdgcn_wmma_f32_16x16x32_f16(false, (a), false, (b), (short)0, (c), false, false)

// ---- TDM: async copy of nelem contiguous f16 elements global -> LDS ---------
// Builds a D# per ISA 8.3/8.4: count=1, type=2 (image), data_size=1 (2 bytes),
// 2D tile: tile_dim0 = tensor_dim0 = nelem, tile_dim1 = tensor_dim1 = 1.
// Groups 2/3 (and trailing group arg) zero. Tracked by TENSORcnt.
// This toolchain exposes the 6-arg builtin:
//   (uint32x4 g0, int32x8 g1, int32x4 g2, int32x4 g3, int32x8 gx, i32 cpol)
static __device__ __forceinline__ void tdm_load_f16(const _Float16* gsrc,
                                                    unsigned lds_off,
                                                    unsigned nelem) {
  const unsigned long long ga = (unsigned long long)(uintptr_t)gsrc;
  v4u g0;
  g0[0] = 1u;                                   // count=1, user mode
  g0[1] = lds_off;                              // lds_addr (bytes)
  g0[2] = (unsigned)(ga & 0xffffffffu);         // global_addr[31:0]
  g0[3] = (unsigned)((ga >> 32) & 0x01ffffffu)  // global_addr[56:32]
          | (2u << 30);                         // type = 2
  v8i g1;
  g1[0] = (1 << 16);                            // data_size=1 (2B); mask=0
  g1[1] = (int)((nelem & 0xffffu) << 16);       // tensor_dim0[15:0]
  g1[2] = (int)((nelem >> 16) & 0xffffu)        // tensor_dim0[31:16]
          | (1 << 16);                          // tensor_dim1 = 1
  g1[3] = (int)((nelem & 0xffffu) << 16);       // tile_dim0 = nelem
  g1[4] = 1;                                    // tile_dim1 = 1 (tile_dim2=0)
  g1[5] = (int)nelem;                           // tensor_dim0_stride[31:0]
  g1[6] = 0;                                    // stride0[47:32], stride1[15:0]
  g1[7] = 0;                                    // stride1[47:16]
  const v4i gz4 = {0, 0, 0, 0};
  const v8i gz8 = {0, 0, 0, 0, 0, 0, 0, 0};
  __builtin_amdgcn_tensor_load_to_lds(g0, g1, gz4, gz4, gz8, 0);
}

// ============================================================================
// Fused per-instance attention. grid.x = 2*B*N (head instances then tail).
// head (b,i): X[p][d] = g[b,i,p,d] -> gh[b,i,q,d]
// tail (b,j): X[p][d] = g[b,p,j,d] -> gt[b,q,j,d]
// ============================================================================
__launch_bounds__(256, 1)
__global__ void attn_kernel(const float* __restrict__ g,
                            const float* __restrict__ Wq_h, const float* __restrict__ Wk_h,
                            const float* __restrict__ Wv_h,
                            const float* __restrict__ Wq_t, const float* __restrict__ Wk_t,
                            const float* __restrict__ Wv_t,
                            _Float16* __restrict__ gh, _Float16* __restrict__ gt) {
  __shared__ _Float16 Xs[NN * DD];    // 64 KB
  __shared__ _Float16 Ks[NN * DD];    // 64 KB
  __shared__ _Float16 Vs[NN * DD];    // 64 KB
  __shared__ _Float16 Wb[DD * DD];    // 32 KB (one weight at a time; Wq resident last)
  __shared__ _Float16 Qb[32 * DD];    //  8 KB
  __shared__ float    Sb[32 * NN];    // 32 KB
  __shared__ _Float16 Pb[32 * NN];    // 16 KB   total 280 KB < 320 KB/WGP

  const int inst = blockIdx.x;
  const int tail = inst >= BB * NN;
  const int bi   = tail ? inst - BB * NN : inst;
  const int b    = bi / NN;
  const int m    = bi % NN;            // i for head, j for tail
  const int tid  = threadIdx.x;
  const int wv   = tid >> 5;
  const int lane = tid & 31;
  const int cn   = lane & 15, ch = lane >> 4;   // C/D-matrix lane coords

  const float* Wq = tail ? Wq_t : Wq_h;
  const float* Wk = tail ? Wk_t : Wk_h;
  const float* Wv = tail ? Wv_t : Wv_h;

  const size_t baseH = ((size_t)(b * NN + m)) * NN * DD;

  // ---- load X (256x128 f32 -> f16 LDS), float4 vectorized -------------------
#pragma unroll
  for (int it = 0; it < 32; ++it) {
    const int idx4 = it * 256 + tid;       // float4 index; 32 per row
    const int p    = idx4 >> 5;
    const int d4   = idx4 & 31;
    const float* rowp = tail ? (g + ((size_t)(b * NN + p) * NN + m) * DD)
                             : (g + baseH + (size_t)p * DD);
    const float4 x = ((const float4*)rowp)[d4];
    const int o = p * DD + d4 * 4;
    Xs[o + 0] = (_Float16)x.x; Xs[o + 1] = (_Float16)x.y;
    Xs[o + 2] = (_Float16)x.z; Xs[o + 3] = (_Float16)x.w;
  }

  // ---- projection: OUT[p][e] = sum_d Xs[p][d] * W[e][d] ---------------------
  auto project = [&](const float* Wg, _Float16* OUT) {
    __syncthreads();                       // prior Wb readers done; Xs visible
#pragma unroll
    for (int it = 0; it < 16; ++it) {      // 128*128 f32 = 4096 float4
      const int i4 = it * 256 + tid;
      const float4 w = ((const float4*)Wg)[i4];
      const int o = i4 * 4;
      Wb[o + 0] = (_Float16)w.x; Wb[o + 1] = (_Float16)w.y;
      Wb[o + 2] = (_Float16)w.z; Wb[o + 3] = (_Float16)w.w;
    }
    __syncthreads();
    // 16 M-tiles x 8 N-tiles = 128 positions, 16 per wave
#pragma unroll
    for (int t = 0; t < 16; ++t) {
      const int pos = wv * 16 + t;
      const int mi = pos >> 3, ni = pos & 7;
      v8f c = {};
#pragma unroll
      for (int kk = 0; kk < 4; ++kk) {
        const v16h a  = lda (Xs + (mi * 16) * DD + kk * 32, DD);
        const v16h bt = ldbt(Wb + (ni * 16) * DD + kk * 32, DD);
        WMMA_F16(c, a, bt);
      }
#pragma unroll
      for (int r = 0; r < 8; ++r)
        OUT[(mi * 16 + r + 8 * ch) * DD + ni * 16 + cn] = (_Float16)c[r];
    }
  };

  project(Wk, Ks);
  project(Wv, Vs);

  // ---- Wq resident in Wb for all row blocks ---------------------------------
  __syncthreads();
#pragma unroll
  for (int it = 0; it < 16; ++it) {
    const int i4 = it * 256 + tid;
    const float4 w = ((const float4*)Wq)[i4];
    const int o = i4 * 4;
    Wb[o + 0] = (_Float16)w.x; Wb[o + 1] = (_Float16)w.y;
    Wb[o + 2] = (_Float16)w.z; Wb[o + 3] = (_Float16)w.w;
  }
  __syncthreads();

  const float scale = 0.08838834764831845f;  // 1/sqrt(128)

  // ---- attention over 8 blocks of 32 query rows -----------------------------
  for (int blk = 0; blk < 8; ++blk) {
    const int row0 = blk * 32;

    // Q block: 2x8 tiles, 2 per wave, K=128 -> 4 chained WMMA
#pragma unroll
    for (int t = 0; t < 2; ++t) {
      const int pos = wv * 2 + t;
      const int mi = pos >> 3, ni = pos & 7;
      v8f c = {};
#pragma unroll
      for (int kk = 0; kk < 4; ++kk) {
        const v16h a  = lda (Xs + (row0 + mi * 16) * DD + kk * 32, DD);
        const v16h bt = ldbt(Wb + (ni * 16) * DD + kk * 32, DD);
        WMMA_F16(c, a, bt);
      }
#pragma unroll
      for (int r = 0; r < 8; ++r)
        Qb[(mi * 16 + r + 8 * ch) * DD + ni * 16 + cn] = (_Float16)c[r];
    }
    __syncthreads();

    // S = Q*K^T*scale : 2x16 tiles, 4 per wave
#pragma unroll
    for (int t = 0; t < 4; ++t) {
      const int pos = wv * 4 + t;
      const int mi = pos >> 4, ni = pos & 15;
      v8f c = {};
#pragma unroll
      for (int kk = 0; kk < 4; ++kk) {
        const v16h a  = lda (Qb + (mi * 16) * DD + kk * 32, DD);
        const v16h bt = ldbt(Ks + (ni * 16) * DD + kk * 32, DD);
        WMMA_F16(c, a, bt);
      }
#pragma unroll
      for (int r = 0; r < 8; ++r)
        Sb[(mi * 16 + r + 8 * ch) * NN + ni * 16 + cn] = c[r] * scale;
    }
    __syncthreads();

    // masked softmax: 8 lanes per row, 32 cols each
    {
      const int r    = tid >> 3;
      const int part = tid & 7;
      const int grow = row0 + r;            // diagonal column excluded
      float ev[32];
      float mx = -1e30f;
      const float* srow = Sb + r * NN + part * 32;
#pragma unroll
      for (int c0 = 0; c0 < 32; ++c0) {
        const int col = part * 32 + c0;
        const float v = (col == grow) ? -1e30f : srow[c0];
        ev[c0] = v;
        mx = fmaxf(mx, v);
      }
#pragma unroll
      for (int off = 1; off < 8; off <<= 1) mx = fmaxf(mx, __shfl_xor(mx, off, 8));
      float sum = 0.0f;
#pragma unroll
      for (int c0 = 0; c0 < 32; ++c0) {
        const float e = __expf(ev[c0] - mx);
        ev[c0] = e;
        sum += e;
      }
#pragma unroll
      for (int off = 1; off < 8; off <<= 1) sum += __shfl_xor(sum, off, 8);
      const float inv = 1.0f / sum;
      _Float16* prow = Pb + r * NN + part * 32;
#pragma unroll
      for (int c0 = 0; c0 < 32; ++c0) prow[c0] = (_Float16)(ev[c0] * inv);
    }
    __syncthreads();

    // O = P*V : 2x8 tiles, 2 per wave, K=256 -> 8 chained WMMA
#pragma unroll
    for (int t = 0; t < 2; ++t) {
      const int pos = wv * 2 + t;
      const int mi = pos >> 3, ni = pos & 7;
      v8f c = {};
#pragma unroll
      for (int kk = 0; kk < 8; ++kk) {
        const v16h a  = lda(Pb + (mi * 16) * NN + kk * 32, NN);
        const v16h bb = ldb(Vs + (kk * 32) * DD + ni * 16, DD);
        WMMA_F16(c, a, bb);
      }
      _Float16* dst = tail ? gt : gh;
#pragma unroll
      for (int r = 0; r < 8; ++r) {
        const int q = row0 + mi * 16 + r + 8 * ch;
        const int d = ni * 16 + cn;
        const size_t off = tail ? (((size_t)(b * NN + q) * NN + m) * DD + d)
                                : (baseH + (size_t)q * DD + d);
        dst[off] = (_Float16)c[r];
      }
    }
    __syncthreads();
  }
}

// ============================================================================
// Fused MLP: out = silu(concat[g,gh,gt] @ W1^T + b1) @ W2^T + b2
// grid.x = B*N*N/128; 128 flat rows per block.
// gh/gt 32KB slices are contiguous f16 -> staged to LDS by the Tensor Data
// Mover (async, TENSORcnt), overlapped with W1/W2/g loads + f16 conversion.
// ============================================================================
__launch_bounds__(256, 1)
__global__ void mlp_kernel(const float* __restrict__ g,
                           const _Float16* __restrict__ gh, const _Float16* __restrict__ gt,
                           const float* __restrict__ W1, const float* __restrict__ b1,
                           const float* __restrict__ W2, const float* __restrict__ b2,
                           float* __restrict__ out) {
  __shared__ _Float16 Gs [128 * DD];   // 32 KB  g slice (f32 -> f16)
  __shared__ _Float16 GHs[128 * DD];   // 32 KB  head slice (TDM)
  __shared__ _Float16 GTs[128 * DD];   // 32 KB  tail slice (TDM)
  __shared__ _Float16 W1s[128 * 384];  // 96 KB  W1[e][c]
  __shared__ _Float16 W2s[128 * 128];  // 32 KB  W2[e][d]
  __shared__ _Float16 Hs [128 * 128];  // 32 KB  hidden    total 256 KB

  const int tid = threadIdx.x;
  const int wv = tid >> 5, lane = tid & 31;
  const int cn = lane & 15, ch = lane >> 4;
  const size_t row0 = (size_t)blockIdx.x * 128;

  // Kick off the two TDM copies first (wave 0 issues; EXEC-independent DMA).
  if (wv == 0) {
    tdm_load_f16(gh + row0 * DD, (unsigned)(uintptr_t)GHs, 128 * DD);
    tdm_load_f16(gt + row0 * DD, (unsigned)(uintptr_t)GTs, 128 * DD);
  }

  // W1: 12288 float4 -> 48 per thread
#pragma unroll
  for (int it = 0; it < 48; ++it) {
    const int i4 = it * 256 + tid;
    const float4 w = ((const float4*)W1)[i4];
    const int o = i4 * 4;
    W1s[o + 0] = (_Float16)w.x; W1s[o + 1] = (_Float16)w.y;
    W1s[o + 2] = (_Float16)w.z; W1s[o + 3] = (_Float16)w.w;
  }
  // W2: 4096 float4 -> 16 per thread
#pragma unroll
  for (int it = 0; it < 16; ++it) {
    const int i4 = it * 256 + tid;
    const float4 w = ((const float4*)W2)[i4];
    const int o = i4 * 4;
    W2s[o + 0] = (_Float16)w.x; W2s[o + 1] = (_Float16)w.y;
    W2s[o + 2] = (_Float16)w.z; W2s[o + 3] = (_Float16)w.w;
  }
  // g slice: 4096 float4 -> 16 per thread
#pragma unroll
  for (int it = 0; it < 16; ++it) {
    const int i4 = it * 256 + tid;
    const int r = i4 >> 5, d4 = i4 & 31;
    const float4 x = ((const float4*)(g + (row0 + r) * DD))[d4];
    const int o = r * DD + d4 * 4;
    Gs[o + 0] = (_Float16)x.x; Gs[o + 1] = (_Float16)x.y;
    Gs[o + 2] = (_Float16)x.z; Gs[o + 3] = (_Float16)x.w;
  }

  // TENSORcnt is per-wave: the issuing wave drains it, then the barrier
  // publishes the TDM-written LDS to the whole workgroup.
  if (wv == 0) __builtin_amdgcn_s_wait_tensorcnt(0);
  __syncthreads();

  // stage 1: H = silu(concat @ W1^T + b1): 8x8 tiles, 8 per wave, K=384 -> 12
#pragma unroll
  for (int t = 0; t < 8; ++t) {
    const int pos = wv * 8 + t;
    const int mi = pos >> 3, ni = pos & 7;
    v8f c = {};
#pragma unroll
    for (int kk = 0; kk < 12; ++kk) {
      const _Float16* asrc = (kk < 4) ? (Gs  + (mi * 16) * DD + kk * 32)
                           : (kk < 8) ? (GHs + (mi * 16) * DD + (kk - 4) * 32)
                                      : (GTs + (mi * 16) * DD + (kk - 8) * 32);
      const v16h a  = lda (asrc, DD);
      const v16h bt = ldbt(W1s + (ni * 16) * 384 + kk * 32, 384);
      WMMA_F16(c, a, bt);
    }
#pragma unroll
    for (int r = 0; r < 8; ++r) {
      const int e = ni * 16 + cn;
      const float x = c[r] + b1[e];
      const float s = x / (1.0f + __expf(-x));
      Hs[(mi * 16 + r + 8 * ch) * DD + e] = (_Float16)s;
    }
  }
  __syncthreads();

  // stage 2: out = Hs*W2^T + b2: 8x8 tiles, 8 per wave, K=128 -> 4 chained
#pragma unroll
  for (int t = 0; t < 8; ++t) {
    const int pos = wv * 8 + t;
    const int mi = pos >> 3, ni = pos & 7;
    v8f c = {};
#pragma unroll
    for (int kk = 0; kk < 4; ++kk) {
      const v16h a  = lda (Hs  + (mi * 16) * DD + kk * 32, DD);
      const v16h bt = ldbt(W2s + (ni * 16) * DD + kk * 32, DD);
      WMMA_F16(c, a, bt);
    }
#pragma unroll
    for (int r = 0; r < 8; ++r) {
      const int e = ni * 16 + cn;
      out[(row0 + mi * 16 + r + 8 * ch) * DD + e] = c[r] + b2[e];
    }
  }
}

extern "C" void kernel_launch(void* const* d_in, const int* in_sizes, int n_in,
                              void* d_out, int out_size, void* d_ws, size_t ws_size,
                              hipStream_t stream) {
  const float* g    = (const float*)d_in[0];
  const float* Wq_h = (const float*)d_in[1];
  const float* Wk_h = (const float*)d_in[2];
  const float* Wv_h = (const float*)d_in[3];
  const float* Wq_t = (const float*)d_in[4];
  const float* Wk_t = (const float*)d_in[5];
  const float* Wv_t = (const float*)d_in[6];
  const float* W1   = (const float*)d_in[7];
  const float* b1   = (const float*)d_in[8];
  const float* W2   = (const float*)d_in[9];
  const float* b2   = (const float*)d_in[10];

  _Float16* gh = (_Float16*)d_ws;                       // (B,N,N,D) f16
  _Float16* gt = gh + (size_t)BB * NN * NN * DD;        // (B,N,N,D) f16

  attn_kernel<<<2 * BB * NN, 256, 0, stream>>>(g, Wq_h, Wk_h, Wv_h,
                                               Wq_t, Wk_t, Wv_t, gh, gt);
  mlp_kernel<<<(BB * NN * NN) / 128, 256, 0, stream>>>(g, gh, gt, W1, b1, W2, b2,
                                                       (float*)d_out);
}
